// SingleMCBlock_56341380989153
// MI455X (gfx1250) — matrix operand order repaired and checked by reading, not compiled, not run on hardware
//
#include <hip/hip_runtime.h>
#include <hip/hip_bf16.h>

typedef __attribute__((ext_vector_type(16))) _Float16 v16h;
typedef __attribute__((ext_vector_type(8)))  _Float16 v8h;
typedef __attribute__((ext_vector_type(8)))  float    v8f;

#define BATCH 16
#define NNODE 512
#define FDIM  256
#define BN    (BATCH * NNODE)   // 8192
#define NCONC 32
#define KM_ITERS 10
#define CC_ITERS 20

// ---------------------------------------------------------------------------
// WMMA core (gfx1250 wave32): one 16x16 f32 tile per wave, K in steps of 32.
// A is (M x K) f16 row-major, BT is (N x K) f16 row-major (i.e. B transposed),
// so every fragment is two contiguous global_load_b128 per operand.
//   A frag lane(m=lane&15, g=lane>>4): halves [16g..16g+15] of row m
//   B frag lane(n=lane&15, g=lane>>4): halves [16g..16g+15] of column n
// ---------------------------------------------------------------------------
union AFrag { v16h v; v8h h[2]; };

template <int KSTEPS>
__device__ __forceinline__ v8f wmma_rowdot(const _Float16* __restrict__ Arow,
                                           const _Float16* __restrict__ BTrow,
                                           int g) {
  v8f acc = {};
#pragma unroll
  for (int kk = 0; kk < KSTEPS; ++kk) {
    AFrag af;
    af.v = *(const v16h*)(Arow + kk * 32 + g * 16);
    v16h bf = *(const v16h*)(BTrow + kk * 32 + g * 16);
    acc = __builtin_amdgcn_wmma_f32_16x16x32_f16(false, af.v, false, bf,
                                                 (short)0, acc, false, false);
  }
  return acc;
}

// ---------------------------------------------------------------------------
// GEMM 1: Out(M x N) = A(M x K) @ B(K x N) with BT stored (N x K). f32 out.
// ---------------------------------------------------------------------------
template <int KSTEPS>
__global__ __launch_bounds__(256) void gemm_xw_h(const _Float16* __restrict__ A,
                                                 const _Float16* __restrict__ BT,
                                                 float* __restrict__ Out,
                                                 int Ntiles, int lda, int ldbt, int ldo) {
  const int lane = threadIdx.x & 31, w = threadIdx.x >> 5;
  const int wid = blockIdx.x * 8 + w;
  const int mt = wid / Ntiles, nt = wid % Ntiles;
  const int r = lane & 15, g = lane >> 4;
  const _Float16* Arow  = A  + (size_t)(mt * 16 + r) * lda;
  const _Float16* BTrow = BT + (size_t)(nt * 16 + r) * ldbt;
  v8f acc = wmma_rowdot<KSTEPS>(Arow, BTrow, g);
#pragma unroll
  for (int v = 0; v < 8; ++v)
    Out[(size_t)(mt * 16 + v + 8 * g) * ldo + nt * 16 + r] = acc[v];
}

// ---------------------------------------------------------------------------
// GEMM 2 (propagation, fused GCN epilogue):
//   out[b,i,f] = relu( dinv[b,i]*( (adj_b @ Hs_b)[i,f] + Htmp[b,i,f]*dinv[b,i] ) + bias[f] )
// adjh: f16 (B,512,512); HsT: f16 (B,256,512) = (dinv_j * Htmp[b,j,f]) transposed.
// ---------------------------------------------------------------------------
__global__ __launch_bounds__(256) void gemm_ah_h(const _Float16* __restrict__ adjh,
                                                 const _Float16* __restrict__ HsT,
                                                 const float* __restrict__ Htmp,
                                                 const float* __restrict__ dinv,
                                                 const float* __restrict__ bias,
                                                 float* OutF32, _Float16* OutF16) {
  const int lane = threadIdx.x & 31, w = threadIdx.x >> 5;
  const int wid = blockIdx.x * 8 + w;
  const int b = wid >> 9, t = wid & 511;     // 32*16 tiles per batch
  const int mt = t >> 4, nt = t & 15;
  const int r = lane & 15, g = lane >> 4;
  const _Float16* Arow  = adjh + (size_t)b * NNODE * NNODE + (size_t)(mt * 16 + r) * NNODE;
  const _Float16* BTrow = HsT  + (size_t)b * FDIM * NNODE + (size_t)(nt * 16 + r) * NNODE;
  __builtin_prefetch(Htmp + ((size_t)b * NNODE + mt * 16) * FDIM + nt * 16, 0, 1);
  v8f acc = wmma_rowdot<16>(Arow, BTrow, g);    // K = 512
#pragma unroll
  for (int v = 0; v < 8; ++v) {
    const int m = mt * 16 + v + 8 * g;
    const size_t o = ((size_t)b * NNODE + m) * FDIM + nt * 16 + r;
    const float di = dinv[b * NNODE + m];
    const float val = fmaxf(di * (acc[v] + Htmp[o] * di) + bias[nt * 16 + r], 0.0f);
    if (OutF32) OutF32[o] = val;
    if (OutF16) OutF16[o] = (_Float16)val;
  }
}

// ---------------------------------------------------------------------------
// GEMM 3 (kmeans distances): D2[p,c] = p2[p] - 2*(pts@cent^T)[p,c] + c2[c]
// ptsh: f16 (8192 x 256); centh: f16 (32 x 256) == BT directly.
// ---------------------------------------------------------------------------
__global__ __launch_bounds__(256) void gemm_dot_h(const _Float16* __restrict__ ptsh,
                                                  const _Float16* __restrict__ centh,
                                                  const float* __restrict__ p2,
                                                  const float* __restrict__ c2,
                                                  float* __restrict__ D2) {
  const int lane = threadIdx.x & 31, w = threadIdx.x >> 5;
  const int wid = blockIdx.x * 8 + w;
  const int mt = wid >> 1, nt = wid & 1;    // 512 x 2 tiles
  const int r = lane & 15, g = lane >> 4;
  const _Float16* Arow  = ptsh  + (size_t)(mt * 16 + r) * FDIM;
  const _Float16* BTrow = centh + (size_t)(nt * 16 + r) * FDIM;
  v8f acc = wmma_rowdot<8>(Arow, BTrow, g);   // K = 256
#pragma unroll
  for (int v = 0; v < 8; ++v) {
    const int p = mt * 16 + v + 8 * g;
    const int c = nt * 16 + r;
    D2[(size_t)p * NCONC + c] = p2[p] - 2.0f * acc[v] + c2[c];
  }
}

// ---------------------------------------------------------------------------
// conversion / elementwise kernels
// ---------------------------------------------------------------------------
__global__ void conv_h_kernel(const float* __restrict__ src, _Float16* __restrict__ dst, int n) {
  int i = blockIdx.x * blockDim.x + threadIdx.x;
  if (i < n) dst[i] = (_Float16)src[i];
}

// WT[n*256+k] = W[k*256+n]  (f16)
__global__ void transW_kernel(const float* __restrict__ W, _Float16* __restrict__ WT) {
  int i = blockIdx.x * blockDim.x + threadIdx.x;
  if (i >= FDIM * FDIM) return;
  const int n = i >> 8, k = i & 255;
  WT[i] = (_Float16)W[k * FDIM + n];
}

// HsT[b][f*512 + j] = f16( Htmp[b][j*256+f] * dinv[b*512+j] )
__global__ void scaleT_kernel(const float* __restrict__ Htmp, const float* __restrict__ dinv,
                              _Float16* __restrict__ HsT) {
  int idx = blockIdx.x * blockDim.x + threadIdx.x;
  if (idx >= BN * FDIM) return;
  const int b = idx >> 17;              // 512*256 = 2^17
  const int rem = idx & 131071;
  const int f = rem >> 9, j = rem & 511;
  HsT[idx] = (_Float16)(Htmp[((size_t)(b * NNODE + j)) * FDIM + f] * dinv[b * NNODE + j]);
}

__global__ void zero_kernel(float* p, int n) {
  int i = blockIdx.x * blockDim.x + threadIdx.x;
  if (i < n) p[i] = 0.0f;
}

__global__ __launch_bounds__(256) void deg_kernel(const float* __restrict__ adj,
                                                  float* __restrict__ dinv) {
  const int wid = (blockIdx.x * blockDim.x + threadIdx.x) >> 5;  // row 0..8191
  const int lane = threadIdx.x & 31;
  const float* row = adj + (size_t)wid * NNODE;
  float s = 0.0f;
  for (int t = lane; t < NNODE; t += 32) s += row[t];
#pragma unroll
  for (int m = 16; m >= 1; m >>= 1) s += __shfl_xor(s, m, 32);
  if (lane == 0) dinv[wid] = rsqrtf(fmaxf(s + 1.0f, 1.0f));
}

__global__ void rownorm_kernel(const float* __restrict__ pts, float* __restrict__ out, int n) {
  int p = blockIdx.x * blockDim.x + threadIdx.x;
  if (p >= n) return;
  const float* r = pts + (size_t)p * FDIM;
  float s = 0.0f;
  for (int i = 0; i < FDIM; ++i) s += r[i] * r[i];
  out[p] = s;
}

__global__ void copy_kernel(const float* src, float* dst, int n) {
  int i = blockIdx.x * blockDim.x + threadIdx.x;
  if (i < n) dst[i] = src[i];
}

__global__ void argmin_kernel(const float* __restrict__ D2, int* __restrict__ assign,
                              float* conceptsF) {
  int p = blockIdx.x * blockDim.x + threadIdx.x;
  if (p >= BN) return;
  const float* d = D2 + (size_t)p * NCONC;
  float best = d[0]; int bi = 0;
#pragma unroll
  for (int c = 1; c < NCONC; ++c) {
    float v = d[c];
    if (v < best) { best = v; bi = c; }
  }
  assign[p] = bi;
  if (conceptsF) conceptsF[p] = (float)bi;
}

__global__ void accum_cnt_kernel(const int* __restrict__ assign, float* __restrict__ cnt) {
  int p = blockIdx.x * blockDim.x + threadIdx.x;
  if (p < BN) atomicAdd(&cnt[assign[p]], 1.0f);
}

__global__ void accum_sums_kernel(const float* __restrict__ pts, const int* __restrict__ assign,
                                  float* __restrict__ sums) {
  int i = blockIdx.x * blockDim.x + threadIdx.x;
  if (i >= BN * FDIM) return;
  atomicAdd(&sums[assign[i / FDIM] * FDIM + (i % FDIM)], pts[i]);
}

__global__ void update_cent_kernel(const float* __restrict__ sums, const float* __restrict__ cnt,
                                   float* __restrict__ cent) {
  int i = blockIdx.x * blockDim.x + threadIdx.x;
  if (i >= NCONC * FDIM) return;
  float n = cnt[i / FDIM];
  if (n > 0.0f) cent[i] = sums[i] / fmaxf(n, 1.0f);
}

// ---------------------------------------------------------------------------
// Connected components (label propagation), fully LDS-resident per graph.
// ---------------------------------------------------------------------------
__global__ __launch_bounds__(512) void cc_kernel(const float* __restrict__ adj,
                                                 const int* __restrict__ concepts,
                                                 int* __restrict__ assignSeg,
                                                 float* __restrict__ maskF) {
  __shared__ unsigned bits[NNODE][16];
  __shared__ int conc[NNODE];
  __shared__ int lab[NNODE];
  __shared__ int ids[NNODE];
  const int b = blockIdx.x;
  const int i = threadIdx.x;

  conc[i] = concepts[b * NNODE + i];
  __syncthreads();

  const float* arow = adj + (size_t)b * NNODE * NNODE + (size_t)i * NNODE;
  const int ci = conc[i];
  unsigned w[16];
#pragma unroll
  for (int t = 0; t < 16; ++t) w[t] = 0u;
  for (int j = 0; j < NNODE; ++j)
    if (arow[j] > 0.0f && conc[j] == ci) w[j >> 5] |= 1u << (j & 31);
#pragma unroll
  for (int t = 0; t < 16; ++t) bits[i][t] = w[t];
  lab[i] = i;
  __syncthreads();

  for (int it = 0; it < CC_ITERS; ++it) {
    int m = lab[i];
#pragma unroll
    for (int t = 0; t < 16; ++t) {
      unsigned word = bits[i][t];
      while (word) {
        const int j = (t << 5) + __builtin_ctz(word);
        word &= word - 1;
        const int lj = lab[j];
        m = lj < m ? lj : m;
      }
    }
    __syncthreads();
    lab[i] = m;
    __syncthreads();
#pragma unroll
    for (int rr = 0; rr < 2; ++rr) {   // lab = lab[lab], twice
      const int a1 = lab[lab[i]];
      __syncthreads();
      lab[i] = a1;
      __syncthreads();
    }
  }

  ids[i] = (lab[i] == i) ? 1 : 0;
  __syncthreads();
  for (int off = 1; off < NNODE; off <<= 1) {
    const int add = (i >= off) ? ids[i - off] : 0;
    __syncthreads();
    ids[i] += add;
    __syncthreads();
  }
  assignSeg[b * NNODE + i] = ids[lab[i]];
  maskF[b * NNODE + i] = (i < ids[NNODE - 1]) ? 1.0f : 0.0f;
}

// ---------------------------------------------------------------------------
// Segment pooling
// ---------------------------------------------------------------------------
__global__ void segsum_kernel(const float* __restrict__ h, const int* __restrict__ assign,
                              float* __restrict__ xnew) {
  int idx = blockIdx.x * blockDim.x + threadIdx.x;
  if (idx >= BN * FDIM) return;
  const int f = idx % FDIM;
  const int r = idx / FDIM;
  const int i = r % NNODE, b = r / NNODE;
  const int s = assign[b * NNODE + i];
  if (s >= 1) atomicAdd(&xnew[((size_t)(b * NNODE + (s - 1))) * FDIM + f], h[idx]);
}

__global__ void segmax1_kernel(const float* __restrict__ adj, const int* __restrict__ assign,
                               float* __restrict__ adj_r) {
  long long idx = (long long)blockIdx.x * blockDim.x + threadIdx.x;
  if (idx >= (long long)BATCH * NNODE * NNODE) return;
  const int j = (int)(idx % NNODE);
  const long long r = idx / NNODE;
  const int i = (int)(r % NNODE), b = (int)(r / NNODE);
  const float v = adj[idx];
  if (v > 0.0f) {
    const int s = assign[b * NNODE + i];
    if (s >= 1)
      atomicMax((unsigned*)&adj_r[((size_t)(b * NNODE + (s - 1))) * NNODE + j],
                __float_as_uint(v));
  }
}

__global__ void segmax2_kernel(const float* __restrict__ adj_r, const int* __restrict__ assign,
                               float* __restrict__ adjnew) {
  long long idx = (long long)blockIdx.x * blockDim.x + threadIdx.x;
  if (idx >= (long long)BATCH * NNODE * NNODE) return;
  const int j = (int)(idx % NNODE);
  const long long r = idx / NNODE;
  const int s = (int)(r % NNODE), b = (int)(r / NNODE);
  const float v = adj_r[idx];
  if (v > 0.0f) {
    const int t = assign[b * NNODE + j];
    if (t >= 1)
      atomicMax((unsigned*)&adjnew[((size_t)(b * NNODE + s)) * NNODE + (t - 1)],
                __float_as_uint(v));
  }
}

// ---------------------------------------------------------------------------
// host launcher
// ---------------------------------------------------------------------------
extern "C" void kernel_launch(void* const* d_in, const int* in_sizes, int n_in,
                              void* d_out, int out_size, void* d_ws, size_t ws_size,
                              hipStream_t stream) {
  (void)in_sizes; (void)n_in; (void)out_size; (void)ws_size;
  const float* x   = (const float*)d_in[0];
  const float* adj = (const float*)d_in[1];
  const float* W1  = (const float*)d_in[3];
  const float* b1  = (const float*)d_in[4];
  const float* W2  = (const float*)d_in[5];
  const float* b2  = (const float*)d_in[6];

  // ---- output layout (flattened tuple) ----
  float* out       = (float*)d_out;
  float* xnew      = out;                                     // 16*512*256
  float* adjnew    = out + BATCH * NNODE * FDIM;              // 16*512*512
  float* conceptsF = adjnew + (size_t)BATCH * NNODE * NNODE;  // 16*512
  float* Hout      = conceptsF + BN;                          // 16*512*256 (h)
  float* maskF     = Hout + BATCH * NNODE * FDIM;             // 16*512

  // ---- workspace carve (16-byte+ aligned chunks, sizes are powers of two) ----
  char* wsb = (char*)d_ws;
  // aliased block first: Htmp(f32 8MB) + HsT(f16 4MB) + H1h(f16 4MB) == adj_r(16MB)
  float*    Htmp  = (float*)wsb;                wsb += (size_t)BN * FDIM * 4;
  _Float16* HsT   = (_Float16*)wsb;             wsb += (size_t)BN * FDIM * 2;
  _Float16* H1h   = (_Float16*)wsb;             wsb += (size_t)BN * FDIM * 2;
  float*    adj_r = Htmp;                       // alias, 16 MB
  _Float16* Xh    = (_Float16*)wsb;             wsb += (size_t)BN * FDIM * 2;
  _Float16* adjh  = (_Float16*)wsb;             wsb += (size_t)BATCH * NNODE * NNODE * 2;
  _Float16* Houth = (_Float16*)wsb;             wsb += (size_t)BN * FDIM * 2;
  _Float16* W1T   = (_Float16*)wsb;             wsb += (size_t)FDIM * FDIM * 2;
  _Float16* W2T   = (_Float16*)wsb;             wsb += (size_t)FDIM * FDIM * 2;
  _Float16* centh = (_Float16*)wsb;             wsb += (size_t)NCONC * FDIM * 2;
  float*    dinv  = (float*)wsb;                wsb += (size_t)BN * 4;
  float*    cent  = (float*)wsb;                wsb += (size_t)NCONC * FDIM * 4;
  float*    sums  = (float*)wsb;                wsb += (size_t)NCONC * FDIM * 4;
  float*    cnt   = (float*)wsb;                wsb += 32 * 4;   // contiguous after sums
  float*    p2    = (float*)wsb;                wsb += (size_t)BN * 4;
  float*    c2    = (float*)wsb;                wsb += 32 * 4;
  float*    D2    = (float*)wsb;                wsb += (size_t)BN * NCONC * 4;
  int*      assignK   = (int*)wsb;              wsb += (size_t)BN * 4;
  int*      conceptsI = (int*)wsb;              wsb += (size_t)BN * 4;
  int*      assignSeg = (int*)wsb;              wsb += (size_t)BN * 4;

  const int TB = 256;
  const int elemHN = BN * FDIM;                                // 2M
  const long long elemAdj = (long long)BATCH * NNODE * NNODE;  // 4M

  // 1) degree + f16 conversions
  deg_kernel<<<BN * 32 / TB, TB, 0, stream>>>(adj, dinv);
  conv_h_kernel<<<(elemHN + TB - 1) / TB, TB, 0, stream>>>(x, Xh, elemHN);
  conv_h_kernel<<<(int)((elemAdj + TB - 1) / TB), TB, 0, stream>>>(adj, adjh, (int)elemAdj);
  transW_kernel<<<(FDIM * FDIM + TB - 1) / TB, TB, 0, stream>>>(W1, W1T);
  transW_kernel<<<(FDIM * FDIM + TB - 1) / TB, TB, 0, stream>>>(W2, W2T);

  // 2) layer 1
  gemm_xw_h<8><<<(BN / 16) * (FDIM / 16) / 8, TB, 0, stream>>>(Xh, W1T, Htmp,
      FDIM / 16, FDIM, FDIM, FDIM);
  scaleT_kernel<<<(elemHN + TB - 1) / TB, TB, 0, stream>>>(Htmp, dinv, HsT);
  gemm_ah_h<<<BATCH * 512 / 8, TB, 0, stream>>>(adjh, HsT, Htmp, dinv, b1,
                                                nullptr, H1h);

  // 3) layer 2 -> Hout (f32 output h) + Houth (f16 pts)
  gemm_xw_h<8><<<(BN / 16) * (FDIM / 16) / 8, TB, 0, stream>>>(H1h, W2T, Htmp,
      FDIM / 16, FDIM, FDIM, FDIM);
  scaleT_kernel<<<(elemHN + TB - 1) / TB, TB, 0, stream>>>(Htmp, dinv, HsT);
  gemm_ah_h<<<BATCH * 512 / 8, TB, 0, stream>>>(adjh, HsT, Htmp, dinv, b2,
                                                Hout, Houth);

  // 4) kmeans on pts
  copy_kernel<<<(NCONC * FDIM + TB - 1) / TB, TB, 0, stream>>>(Hout, cent, NCONC * FDIM);
  rownorm_kernel<<<(BN + TB - 1) / TB, TB, 0, stream>>>(Hout, p2, BN);
  for (int it = 0; it < KM_ITERS; ++it) {
    conv_h_kernel<<<(NCONC * FDIM + TB - 1) / TB, TB, 0, stream>>>(cent, centh, NCONC * FDIM);
    rownorm_kernel<<<1, 32, 0, stream>>>(cent, c2, NCONC);
    gemm_dot_h<<<(BN / 16) * 2 / 8, TB, 0, stream>>>(Houth, centh, p2, c2, D2);
    argmin_kernel<<<(BN + TB - 1) / TB, TB, 0, stream>>>(D2, assignK, nullptr);
    zero_kernel<<<(NCONC * FDIM + 32 + TB - 1) / TB, TB, 0, stream>>>(sums, NCONC * FDIM + 32);
    accum_cnt_kernel<<<(BN + TB - 1) / TB, TB, 0, stream>>>(assignK, cnt);
    accum_sums_kernel<<<(elemHN + TB - 1) / TB, TB, 0, stream>>>(Hout, assignK, sums);
    update_cent_kernel<<<(NCONC * FDIM + TB - 1) / TB, TB, 0, stream>>>(sums, cnt, cent);
  }
  conv_h_kernel<<<(NCONC * FDIM + TB - 1) / TB, TB, 0, stream>>>(cent, centh, NCONC * FDIM);
  rownorm_kernel<<<1, 32, 0, stream>>>(cent, c2, NCONC);
  gemm_dot_h<<<(BN / 16) * 2 / 8, TB, 0, stream>>>(Houth, centh, p2, c2, D2);
  argmin_kernel<<<(BN + TB - 1) / TB, TB, 0, stream>>>(D2, conceptsI, conceptsF);

  // 5) connected components -> assignSeg, mask_new
  cc_kernel<<<BATCH, NNODE, 0, stream>>>(adj, conceptsI, assignSeg, maskF);

  // 6) pooling (adj_r aliases Htmp/HsT/H1h, all dead now)
  zero_kernel<<<(elemHN + TB - 1) / TB, TB, 0, stream>>>(xnew, elemHN);
  zero_kernel<<<(int)((elemAdj + TB - 1) / TB), TB, 0, stream>>>(adjnew, (int)elemAdj);
  zero_kernel<<<(int)((elemAdj + TB - 1) / TB), TB, 0, stream>>>(adj_r, (int)elemAdj);
  segsum_kernel<<<(elemHN + TB - 1) / TB, TB, 0, stream>>>(Hout, assignSeg, xnew);
  segmax1_kernel<<<(int)((elemAdj + TB - 1) / TB), TB, 0, stream>>>(adj, assignSeg, adj_r);
  segmax2_kernel<<<(int)((elemAdj + TB - 1) / TB), TB, 0, stream>>>(adj_r, assignSeg, adjnew);
}
